// MatchingNetModel_11553462026918
// MI455X (gfx1250) — compile-verified, matrix operand order checked by reading, not run on hardware
//
#include <hip/hip_runtime.h>
#include <hip/hip_bf16.h>

// ---------------------------------------------------------------------------
// MatchingNet forward for MI455X (gfx1250, wave32, WMMA + Tensor Data Mover).
// Heavy GEMMs: v_wmma_f32_16x16x32_bf16, 256x128 block tile, double-buffered
// LDS staged by TDM (tensor_load_to_lds, TENSORcnt) when available.
// ---------------------------------------------------------------------------

typedef __attribute__((ext_vector_type(16))) __bf16 v16bf;
typedef __attribute__((ext_vector_type(8)))  float  v8f;

#define NSET 4096
#define EDIM 512
#define GDIM 256
#define RDIM 1000

#define BM 256
#define BN 128
#define BK 32
#define LDSS 40                    // 32 + 8 pad elems: 80B LDS row stride
#define SZA (BM * LDSS * 2)        // bytes per A tile buffer
#define SZB (BN * LDSS * 2)        // bytes per B tile buffer
#define LDS_A(b) ((unsigned)((b) * (SZA + SZB)))
#define LDS_B(b) ((unsigned)((b) * (SZA + SZB) + SZA))

#if defined(__has_builtin)
#  if __has_builtin(__builtin_amdgcn_tensor_load_to_lds) && \
      __has_builtin(__builtin_amdgcn_s_wait_tensorcnt)
#    define USE_TDM 1
#  endif
#endif
#ifndef USE_TDM
#  define USE_TDM 0
#endif

// The TDM builtin takes only scalar descriptors, so LLVM cannot see that it
// writes LDS. An operand-less "memory" asm is not enough: non-escaping shared
// memory is provably untouchable by it, so loads of the (apparently never-
// stored) array still fold to undef. Passing the LDS pointer INTO the asm
// makes it escape, forcing the compiler to assume the barrier writes it.
#define TDM_LDS_WRITE_BARRIER(p) asm volatile("" :: "r"((void*)(p)) : "memory")

#if USE_TDM
typedef __attribute__((ext_vector_type(4))) unsigned tdm_v4u;
typedef __attribute__((ext_vector_type(8))) int      tdm_v8i;
typedef __attribute__((ext_vector_type(4))) int      tdm_v4i;

// DMA one (tile_rows x 32) bf16 tile of a 2D tensor into LDS at lds_byte_off.
// D# per CDNA5 ISA §8.3/§8.4: data_size=2B; LDS padding of 16B after every
// 64B (one 32-elem row) reproduces the LDSS=40 padded row stride.
__device__ __forceinline__
void tdm_load_tile(unsigned lds_byte_off, const __bf16* gsrc,
                   unsigned tile_rows, unsigned tensor_cols,
                   unsigned tensor_rows, unsigned stride_elems)
{
    unsigned long long ga = (unsigned long long)(const char*)gsrc;

    tdm_v4u g0;
    g0[0] = 1u;                                                  // count=1 (user D#)
    g0[1] = lds_byte_off;                                        // lds_addr
    g0[2] = (unsigned)ga;                                        // global_addr[31:0]
    g0[3] = (unsigned)((ga >> 32) & 0x01ffffffu) | (2u << 30);   // addr[56:32] | type=2

    tdm_v8i g1;
    g1[0] = (int)((1u << 16)        // data_size = 2 bytes
                | (1u << 20)        // pad_enable
                | (3u << 22)        // pad_interval: 16 DWORDs (64B row)
                | (3u << 25));      // pad_amount:   4 DWORDs (16B)
    g1[1] = (int)((tensor_cols & 0xffffu) << 16);                // dim0[15:0]
    g1[2] = (int)((tensor_cols >> 16) |
                  ((tensor_rows & 0xffffu) << 16));              // dim0[31:16], dim1[15:0]
    g1[3] = (int)((tensor_rows >> 16) | ((unsigned)BK << 16));   // dim1[31:16], tile_dim0=32
    g1[4] = (int)(tile_rows & 0xffffu);                          // tile_dim1; tile_dim2=0
    g1[5] = (int)stride_elems;                                   // dim0_stride[31:0]
    g1[6] = 0;
    g1[7] = 0;

    tdm_v4i z4 = {0, 0, 0, 0};
#if __clang_major__ >= 23
    tdm_v8i z8 = {0, 0, 0, 0, 0, 0, 0, 0};
    __builtin_amdgcn_tensor_load_to_lds(g0, g1, z4, z4, z8, 0);
#else
    __builtin_amdgcn_tensor_load_to_lds(g0, g1, z4, z4, 0);
#endif
}
#endif // USE_TDM

// ---------------------------------------------------------------------------
// NT GEMM: C (MxN, f32) = A (MxK, bf16, row-major) * B (NxK, bf16, row-major)^T
// 256 threads = 8 waves (4x2); block tile 256x128; wave tile 64x64 (4x4 wmma).
// ---------------------------------------------------------------------------
__global__ __launch_bounds__(256)
void gemm_nt_bf16(float* __restrict__ C, int ldc,
                  const __bf16* __restrict__ A, int lda,
                  const __bf16* __restrict__ B, int ldb,
                  int M, int N, int K)
{
    __shared__ char smem[2 * (SZA + SZB)];     // double-buffered A/B tiles

    const int tid  = threadIdx.x;
    const int lane = tid & 31;
    const int l16  = lane & 15;
    const int hi   = lane >> 4;                // K-half select per ISA layout

    // Wave id as a provably-scalar value: TENSOR_LOAD_TO_LDS ignores EXEC, so
    // the leader guard below must be a scalar branch (s_cbranch), never an
    // exec-mask region — otherwise every wave would re-issue the DMA.
    const int wave = __builtin_amdgcn_readfirstlane(tid) >> 5;

    const int bm = blockIdx.y * BM;
    const int bn = blockIdx.x * BN;
    const int wm = (wave >> 1) * 64;           // wave M offset inside tile
    const int wn = (wave & 1) * 64;            // wave N offset inside tile

    v8f acc[4][4];
    #pragma unroll
    for (int i = 0; i < 4; ++i)
        #pragma unroll
        for (int j = 0; j < 4; ++j)
            #pragma unroll
            for (int k = 0; k < 8; ++k)
                acc[i][j][k] = 0.0f;

    union FragU { uint4 u[2]; v16bf v; };
    const int ktiles = K / BK;

    // ---- prologue: stage tile 0 into buffer 0 ----
#if USE_TDM
    if (wave == 0) {
        tdm_load_tile(LDS_A(0), A + (size_t)bm * lda, BM, (unsigned)K, (unsigned)M, (unsigned)lda);
        tdm_load_tile(LDS_B(0), B + (size_t)bn * ldb, BN, (unsigned)K, (unsigned)N, (unsigned)ldb);
        __builtin_amdgcn_s_wait_tensorcnt(0);
    }
    TDM_LDS_WRITE_BARRIER(smem);
#else
    {
        __bf16* dA = (__bf16*)(smem + LDS_A(0));
        __bf16* dB = (__bf16*)(smem + LDS_B(0));
        #pragma unroll
        for (int i = 0; i < 4; ++i) {          // A: 256x32 = 1024 x 16B chunks
            int ch = tid + i * 256, row = ch >> 2, cc = (ch & 3) * 8;
            *(uint4*)(dA + row * LDSS + cc) = *(const uint4*)(A + (size_t)(bm + row) * lda + cc);
        }
        #pragma unroll
        for (int i = 0; i < 2; ++i) {          // B: 128x32 = 512 x 16B chunks
            int ch = tid + i * 256, row = ch >> 2, cc = (ch & 3) * 8;
            *(uint4*)(dB + row * LDSS + cc) = *(const uint4*)(B + (size_t)(bn + row) * ldb + cc);
        }
    }
#endif
    __syncthreads();

    for (int kt = 0; kt < ktiles; ++kt) {
        const int cur = kt & 1;
        const int nxt = cur ^ 1;
        const bool more = (kt + 1 < ktiles);

        // ---- issue next tile's fill (overlaps with this tile's WMMAs) ----
#if USE_TDM
        if (wave == 0 && more) {
            const int k1 = (kt + 1) * BK;
            tdm_load_tile(LDS_A(nxt), A + (size_t)bm * lda + k1, BM, (unsigned)K, (unsigned)M, (unsigned)lda);
            tdm_load_tile(LDS_B(nxt), B + (size_t)bn * ldb + k1, BN, (unsigned)K, (unsigned)N, (unsigned)ldb);
        }
        TDM_LDS_WRITE_BARRIER(smem);
#else
        uint4 ra[4], rb[2];
        if (more) {
            const int k1 = (kt + 1) * BK;
            #pragma unroll
            for (int i = 0; i < 4; ++i) {
                int ch = tid + i * 256, row = ch >> 2, cc = (ch & 3) * 8;
                ra[i] = *(const uint4*)(A + (size_t)(bm + row) * lda + k1 + cc);
            }
            #pragma unroll
            for (int i = 0; i < 2; ++i) {
                int ch = tid + i * 256, row = ch >> 2, cc = (ch & 3) * 8;
                rb[i] = *(const uint4*)(B + (size_t)(bn + row) * ldb + k1 + cc);
            }
        }
#endif

        // ---- compute on current buffer ----
        const __bf16* sA = (const __bf16*)(smem + LDS_A(cur));
        const __bf16* sB = (const __bf16*)(smem + LDS_B(cur));

        // fragment loads per documented 16-bit 16x32 layout:
        // lanes 0-15: K {0..7,16..23}; lanes 16-31: K {8..15,24..31}
        v16bf afrag[4], bfrag[4];
        #pragma unroll
        for (int mi = 0; mi < 4; ++mi) {
            FragU f;
            const __bf16* p = sA + (wm + mi * 16 + l16) * LDSS + hi * 8;
            f.u[0] = *(const uint4*)(p);
            f.u[1] = *(const uint4*)(p + 16);
            afrag[mi] = f.v;
        }
        #pragma unroll
        for (int ni = 0; ni < 4; ++ni) {
            FragU f;
            const __bf16* p = sB + (wn + ni * 16 + l16) * LDSS + hi * 8;
            f.u[0] = *(const uint4*)(p);
            f.u[1] = *(const uint4*)(p + 16);
            bfrag[ni] = f.v;
        }
        #pragma unroll
        for (int mi = 0; mi < 4; ++mi)
            #pragma unroll
            for (int ni = 0; ni < 4; ++ni)
                acc[mi][ni] = __builtin_amdgcn_wmma_f32_16x16x32_bf16(
                    false, afrag[mi], false, bfrag[ni],
                    (short)0, acc[mi][ni], false, false);

        // ---- retire next tile's fill, then publish ----
#if USE_TDM
        if (wave == 0 && more) __builtin_amdgcn_s_wait_tensorcnt(0);
        TDM_LDS_WRITE_BARRIER(smem);
#else
        if (more) {
            __bf16* dA = (__bf16*)(smem + LDS_A(nxt));
            __bf16* dB = (__bf16*)(smem + LDS_B(nxt));
            #pragma unroll
            for (int i = 0; i < 4; ++i) {
                int ch = tid + i * 256, row = ch >> 2, cc = (ch & 3) * 8;
                *(uint4*)(dA + row * LDSS + cc) = ra[i];
            }
            #pragma unroll
            for (int i = 0; i < 2; ++i) {
                int ch = tid + i * 256, row = ch >> 2, cc = (ch & 3) * 8;
                *(uint4*)(dB + row * LDSS + cc) = rb[i];
            }
        }
#endif
        __syncthreads();
    }

    // D layout: VGPR v: lanes 0-15 -> M=v, lanes 16-31 -> M=8+v; N = lane%16
    #pragma unroll
    for (int mi = 0; mi < 4; ++mi)
        #pragma unroll
        for (int ni = 0; ni < 4; ++ni) {
            int col     = bn + wn + ni * 16 + l16;
            int rowbase = bm + wm + mi * 16 + hi * 8;
            #pragma unroll
            for (int v = 0; v < 8; ++v)
                C[(size_t)(rowbase + v) * ldc + col] = acc[mi][ni][v];
        }
}

// ---------------------------------------------------------------------------
// helpers / elementwise stages
// ---------------------------------------------------------------------------
__global__ void f32_to_bf16_kernel(const float* __restrict__ in,
                                   __bf16* __restrict__ out, size_t n)
{
    size_t t = (size_t)blockIdx.x * blockDim.x + threadIdx.x;
    if (t < n) out[t] = (__bf16)in[t];
}

__global__ void gather_embed_kernel(const int* __restrict__ idx,
                                    const float* __restrict__ emb,
                                    __bf16* __restrict__ out16,
                                    float* __restrict__ out32, int n_rows)
{
    size_t t = (size_t)blockIdx.x * blockDim.x + threadIdx.x;
    size_t total = (size_t)n_rows * EDIM;
    if (t >= total) return;
    int n = (int)(t >> 9);
    int e = (int)(t & 511);
    float v = emb[(size_t)idx[n] * EDIM + e];
    out16[t] = (__bf16)v;
    if (out32) out32[t] = v;
}

// one LSTM cell pointwise stage (gate order i,f,g,o), nullable extras.
__global__ void lstm_cell_kernel(const float* __restrict__ P,
                                 const float* __restrict__ Q,
                                 const float* __restrict__ bih,
                                 const float* __restrict__ bhh,
                                 const float* __restrict__ c_in,
                                 const float* __restrict__ addx,
                                 int H,
                                 float* __restrict__ c_out,
                                 float* __restrict__ h32, int ldh32,
                                 __bf16* __restrict__ h16, int ldh16,
                                 __bf16* __restrict__ hT,  int ldT)
{
    size_t t = (size_t)blockIdx.x * blockDim.x + threadIdx.x;
    size_t total = (size_t)NSET * H;
    if (t >= total) return;
    int n = (int)(t / H);
    int j = (int)(t % H);
    size_t rb = (size_t)n * 4 * H;

    float gi = P[rb + j]         + bih[j]         + bhh[j];
    float gf = P[rb + H + j]     + bih[H + j]     + bhh[H + j];
    float gg = P[rb + 2 * H + j] + bih[2 * H + j] + bhh[2 * H + j];
    float go = P[rb + 3 * H + j] + bih[3 * H + j] + bhh[3 * H + j];
    if (Q) {
        gi += Q[rb + j]; gf += Q[rb + H + j];
        gg += Q[rb + 2 * H + j]; go += Q[rb + 3 * H + j];
    }
    float si = 1.f / (1.f + __expf(-gi));
    float sf = 1.f / (1.f + __expf(-gf));
    float so = 1.f / (1.f + __expf(-go));
    float c  = si * tanhf(gg) + (c_in ? sf * c_in[t] : 0.f);
    float h  = so * tanhf(c);
    if (addx) h += addx[t];

    if (c_out) c_out[t] = c;
    if (h32) h32[(size_t)n * ldh32 + j] = h;
    if (h16) h16[(size_t)n * ldh16 + j] = (__bf16)h;
    if (hT)  hT[(size_t)j * ldT + n]    = (__bf16)h;
}

// 2-way softmax over batch axis -> bf16 attention weights
__global__ void softmax2_kernel(const float* __restrict__ s,
                                __bf16* __restrict__ a, size_t count)
{
    size_t t = (size_t)blockIdx.x * blockDim.x + threadIdx.x;
    if (t >= count) return;
    float s0 = s[t], s1 = s[count + t];
    float m  = fmaxf(s0, s1);
    float e0 = __expf(s0 - m), e1 = __expf(s1 - m);
    float inv = 1.f / (e0 + e1);
    a[t]         = (__bf16)(e0 * inv);
    a[count + t] = (__bf16)(e1 * inv);
}

// per (b,e): dot / |r|^2 / |g|^2 over the n axis (4096)
__global__ void dotnorm_kernel(const float* __restrict__ r,
                               const float* __restrict__ g,
                               float* __restrict__ dot,
                               float* __restrict__ rn2,
                               float* __restrict__ gn2)
{
    __shared__ float bufd[256], bufr[256], bufg[256];
    int be = blockIdx.x;                 // 0..1023
    int b = be >> 9, e = be & 511;
    const float* rp = r + (size_t)b * NSET * EDIM + e;
    const float* gp = g + (size_t)b * NSET * EDIM + e;
    float sd = 0.f, sr = 0.f, sg = 0.f;
    for (int n = threadIdx.x; n < NSET; n += 256) {
        float rv = rp[(size_t)n * EDIM];
        float gv = gp[(size_t)n * EDIM];
        sd += rv * gv; sr += rv * rv; sg += gv * gv;
    }
    int t = threadIdx.x;
    bufd[t] = sd; bufr[t] = sr; bufg[t] = sg;
    __syncthreads();
    for (int s = 128; s > 0; s >>= 1) {
        if (t < s) { bufd[t] += bufd[t + s]; bufr[t] += bufr[t + s]; bufg[t] += bufg[t + s]; }
        __syncthreads();
    }
    if (t == 0) { dot[be] = bufd[0]; rn2[be] = bufr[0]; gn2[be] = bufg[0]; }
}

// column sums of w_out (R x E) -> wcol (E)
__global__ void wcol_kernel(const float* __restrict__ w, float* __restrict__ wcol)
{
    int e = blockIdx.x * 256 + threadIdx.x;
    if (e >= EDIM) return;
    float s = 0.f;
    for (int i = 0; i < RDIM; ++i) s += w[(size_t)i * EDIM + e];
    wcol[e] = s;
}

__device__ __forceinline__ float block_sum(float v, float* buf)
{
    int t = threadIdx.x;
    buf[t] = v; __syncthreads();
    for (int s = blockDim.x >> 1; s > 0; s >>= 1) {
        if (t < s) buf[t] += buf[t + s];
        __syncthreads();
    }
    float r = buf[0]; __syncthreads();
    return r;
}

__device__ __forceinline__ float block_max(float v, float* buf)
{
    int t = threadIdx.x;
    buf[t] = v; __syncthreads();
    for (int s = blockDim.x >> 1; s > 0; s >>= 1) {
        if (t < s) buf[t] = fmaxf(buf[t], buf[t + s]);
        __syncthreads();
    }
    float r = buf[0]; __syncthreads();
    return r;
}

// cosine -> exp-normalize -> collapsed projection -> label mix -> softmax(R)
__global__ __launch_bounds__(1024)
void finisher_kernel(const float* __restrict__ dot,
                     const float* __restrict__ rn2,
                     const float* __restrict__ gn2,
                     const float* __restrict__ wcol,
                     const float* __restrict__ b_out,
                     const float* __restrict__ labels,
                     float* __restrict__ out)
{
    __shared__ float buf[1024];
    int t = threadIdx.x;                 // 1024 threads == 2*EDIM
    int b = t >> 9, e = t & 511;

    float nr = fmaxf(sqrtf(rn2[t]), 1e-8f);
    float ng = fmaxf(sqrtf(gn2[t]), 1e-8f);
    float c  = dot[t] / (nr * ng);

    float S    = block_sum(__expf(c), buf);                 // sum(exp(cos))
    float val  = c * wcol[e];
    float s0   = block_sum(b == 0 ? val : 0.f, buf);
    float s1   = block_sum(b == 1 ? val : 0.f, buf);
    float bsum = block_sum(t < RDIM ? b_out[t] : 0.f, buf);
    s0 = s0 / S + bsum;
    s1 = s1 / S + bsum;

    float v = -1e30f;
    if (t < RDIM) v = s0 * labels[t] + s1 * labels[RDIM + t];
    float m = block_max(v, buf);
    float Z = block_sum(t < RDIM ? __expf(v - m) : 0.f, buf);
    if (t < RDIM) out[t] = __expf(v - m) / Z;
}

// ---------------------------------------------------------------------------
// launcher
// ---------------------------------------------------------------------------
extern "C" void kernel_launch(void* const* d_in, const int* in_sizes, int n_in,
                              void* d_out, int out_size, void* d_ws, size_t ws_size,
                              hipStream_t stream)
{
    const int*   input      = (const int*)d_in[0];
    const int*   set_inputs = (const int*)d_in[1];
    const float* set_labels = (const float*)d_in[2];
    const float* embedding  = (const float*)d_in[3];
    const float* wih_gf = (const float*)d_in[4];
    const float* whh_gf = (const float*)d_in[5];
    const float* bih_gf = (const float*)d_in[6];
    const float* bhh_gf = (const float*)d_in[7];
    const float* wih_gr = (const float*)d_in[8];
    const float* whh_gr = (const float*)d_in[9];
    const float* bih_gr = (const float*)d_in[10];
    const float* bhh_gr = (const float*)d_in[11];
    const float* wih_f  = (const float*)d_in[12];
    /* d_in[13] whh_f: unused — h==0 for the single f-LSTM step */
    const float* bih_f  = (const float*)d_in[14];
    const float* bhh_f  = (const float*)d_in[15];
    const float* w_out  = (const float*)d_in[16];
    const float* b_out  = (const float*)d_in[17];
    float* out = (float*)d_out;

    char* wsb = (char*)d_ws;
    size_t off = 0;
    auto alloc = [&](size_t bytes) -> char* {
        char* p = wsb + off;
        off += (bytes + 255) & ~(size_t)255;
        return p;
    };

    __bf16* xh      = (__bf16*)alloc((size_t)NSET * EDIM * 2);
    float*  xf      = (float*) alloc((size_t)NSET * EDIM * 4);
    __bf16* exh     = (__bf16*)alloc((size_t)2 * NSET * EDIM * 2);
    __bf16* w16_gf  = (__bf16*)alloc((size_t)1024 * 512 * 2);
    __bf16* w16_gr  = (__bf16*)alloc((size_t)1024 * 512 * 2);
    __bf16* wh16_gf = (__bf16*)alloc((size_t)1024 * 256 * 2);
    __bf16* wh16_gr = (__bf16*)alloc((size_t)1024 * 256 * 2);
    __bf16* w16_f   = (__bf16*)alloc((size_t)2048 * 512 * 2);
    float*  Pgf0    = (float*) alloc((size_t)NSET * 1024 * 4);
    float*  Pgf1    = (float*) alloc((size_t)NSET * 1024 * 4);
    float*  Pgr0    = (float*) alloc((size_t)NSET * 1024 * 4);
    float*  Pgr1    = (float*) alloc((size_t)NSET * 1024 * 4);
    float*  Pf      = (float*) alloc((size_t)NSET * 2048 * 4);
    float*  Qf      = (float*) alloc((size_t)NSET * 1024 * 4);
    float*  Qr      = (float*) alloc((size_t)NSET * 1024 * 4);
    float*  cfb     = (float*) alloc((size_t)NSET * 256 * 4);
    float*  crb     = (float*) alloc((size_t)NSET * 256 * 4);
    float*  g32     = (float*) alloc((size_t)2 * NSET * EDIM * 4);
    __bf16* g16     = (__bf16*)alloc((size_t)2 * NSET * EDIM * 2);
    __bf16* gT16    = (__bf16*)alloc((size_t)2 * EDIM * NSET * 2);
    __bf16* fh16    = (__bf16*)alloc((size_t)NSET * EDIM * 2);
    float*  scores  = (float*) alloc((size_t)2 * NSET * NSET * 4);
    __bf16* a16     = (__bf16*)alloc((size_t)2 * NSET * NSET * 2);
    float*  r32     = (float*) alloc((size_t)2 * NSET * EDIM * 4);
    float*  dotb    = (float*) alloc(1024 * 4);
    float*  rn2b    = (float*) alloc(1024 * 4);
    float*  gn2b    = (float*) alloc(1024 * 4);
    float*  wcol    = (float*) alloc(512 * 4);

    auto cvt = [&](const float* src, __bf16* dst, size_t n) {
        f32_to_bf16_kernel<<<dim3((unsigned)((n + 255) / 256)), 256, 0, stream>>>(src, dst, n);
    };
    cvt(wih_gf, w16_gf, (size_t)1024 * 512);
    cvt(wih_gr, w16_gr, (size_t)1024 * 512);
    cvt(whh_gf, wh16_gf, (size_t)1024 * 256);
    cvt(whh_gr, wh16_gr, (size_t)1024 * 256);
    cvt(wih_f,  w16_f,  (size_t)2048 * 512);

    gather_embed_kernel<<<dim3(NSET * EDIM / 256), 256, 0, stream>>>(input, embedding, xh, xf, NSET);
    gather_embed_kernel<<<dim3(2 * NSET * EDIM / 256), 256, 0, stream>>>(set_inputs, embedding, exh, nullptr, 2 * NSET);

    // phase 1: all input-side GEMMs (independent)
    gemm_nt_bf16<<<dim3(1024 / BN, NSET / BM), 256, 0, stream>>>(Pgf0, 1024, exh, EDIM, w16_gf, EDIM, NSET, 1024, EDIM);
    gemm_nt_bf16<<<dim3(1024 / BN, NSET / BM), 256, 0, stream>>>(Pgf1, 1024, exh + (size_t)NSET * EDIM, EDIM, w16_gf, EDIM, NSET, 1024, EDIM);
    gemm_nt_bf16<<<dim3(1024 / BN, NSET / BM), 256, 0, stream>>>(Pgr1, 1024, exh + (size_t)NSET * EDIM, EDIM, w16_gr, EDIM, NSET, 1024, EDIM);
    gemm_nt_bf16<<<dim3(1024 / BN, NSET / BM), 256, 0, stream>>>(Pgr0, 1024, exh, EDIM, w16_gr, EDIM, NSET, 1024, EDIM);
    gemm_nt_bf16<<<dim3(2048 / BN, NSET / BM), 256, 0, stream>>>(Pf, 2048, xh, EDIM, w16_f, EDIM, NSET, 2048, EDIM);

    // phase 2: cells with h=0 (hf0, hr1) + f-cell (fh = h1 + x)
    lstm_cell_kernel<<<dim3(NSET * GDIM / 256), 256, 0, stream>>>(
        Pgf0, nullptr, bih_gf, bhh_gf, nullptr, nullptr, GDIM,
        cfb, g32, EDIM, g16, EDIM, gT16, NSET);                            // hf0 -> b0 cols[0:256)
    lstm_cell_kernel<<<dim3(NSET * GDIM / 256), 256, 0, stream>>>(
        Pgr1, nullptr, bih_gr, bhh_gr, nullptr, nullptr, GDIM,
        crb, g32 + (size_t)NSET * EDIM + GDIM, EDIM,
        g16 + (size_t)NSET * EDIM + GDIM, EDIM,
        gT16 + (size_t)(EDIM + GDIM) * NSET, NSET);                        // hr1 -> b1 cols[256:512)
    lstm_cell_kernel<<<dim3(NSET * EDIM / 256), 256, 0, stream>>>(
        Pf, nullptr, bih_f, bhh_f, nullptr, xf, EDIM,
        nullptr, nullptr, 0, fh16, EDIM, nullptr, 0);                      // fh bf16

    // phase 3: recurrent GEMMs (A = hidden slices inside g16, lda = EDIM)
    gemm_nt_bf16<<<dim3(1024 / BN, NSET / BM), 256, 0, stream>>>(Qf, 1024, g16, EDIM, wh16_gf, GDIM, NSET, 1024, GDIM);
    gemm_nt_bf16<<<dim3(1024 / BN, NSET / BM), 256, 0, stream>>>(Qr, 1024, g16 + (size_t)NSET * EDIM + GDIM, EDIM, wh16_gr, GDIM, NSET, 1024, GDIM);

    // phase 4: second-step cells (hf1, hr0)
    lstm_cell_kernel<<<dim3(NSET * GDIM / 256), 256, 0, stream>>>(
        Pgf1, Qf, bih_gf, bhh_gf, cfb, nullptr, GDIM,
        nullptr, g32 + (size_t)NSET * EDIM, EDIM,
        g16 + (size_t)NSET * EDIM, EDIM, gT16 + (size_t)EDIM * NSET, NSET); // hf1 -> b1 cols[0:256)
    lstm_cell_kernel<<<dim3(NSET * GDIM / 256), 256, 0, stream>>>(
        Pgr0, Qr, bih_gr, bhh_gr, crb, nullptr, GDIM,
        nullptr, g32 + GDIM, EDIM,
        g16 + GDIM, EDIM, gT16 + (size_t)GDIM * NSET, NSET);                // hr0 -> b0 cols[256:512)

    // phase 5: scores[b] = g_out[b] (N x E) * fh^T  -> (N x N)
    for (int b = 0; b < 2; ++b)
        gemm_nt_bf16<<<dim3(NSET / BN, NSET / BM), 256, 0, stream>>>(
            scores + (size_t)b * NSET * NSET, NSET,
            g16 + (size_t)b * NSET * EDIM, EDIM, fh16, EDIM, NSET, NSET, EDIM);

    // phase 6: 2-way softmax over batch axis
    softmax2_kernel<<<dim3((unsigned)((size_t)NSET * NSET / 256)), 256, 0, stream>>>(
        scores, a16, (size_t)NSET * NSET);

    // phase 7: r[b] = a[b] (N x N) * g_outT[b] (E x N)^T -> (N x E)
    for (int b = 0; b < 2; ++b)
        gemm_nt_bf16<<<dim3(EDIM / BN, NSET / BM), 256, 0, stream>>>(
            r32 + (size_t)b * NSET * EDIM, EDIM,
            a16 + (size_t)b * NSET * NSET, NSET,
            gT16 + (size_t)b * EDIM * NSET, NSET, NSET, EDIM, NSET);

    // phase 8: reductions over n, collapsed projection, final softmax
    dotnorm_kernel<<<dim3(1024), 256, 0, stream>>>(r32, g32, dotb, rn2b, gn2b);
    wcol_kernel<<<dim3(2), 256, 0, stream>>>(w_out, wcol);
    finisher_kernel<<<dim3(1), 1024, 0, stream>>>(dotb, rn2b, gn2b, wcol, b_out, set_labels, out);
}
